// EnhancedEdgeGNN_51127290692283
// MI455X (gfx1250) — compile-verified, matrix operand order checked by reading, not run on hardware
//
#include <hip/hip_runtime.h>
#include <hip/hip_bf16.h>

typedef __attribute__((ext_vector_type(16))) _Float16 v16h;
typedef __attribute__((ext_vector_type(8)))  float    v8f;
typedef _Float16 half_t;

#define HID 64

// ---------------------------------------------------------------------------
// small utility kernels
// ---------------------------------------------------------------------------
__global__ void k_zero_u32(unsigned* __restrict__ p, long long n) {
  long long i = (long long)blockIdx.x * blockDim.x + threadIdx.x;
  if (i < n) p[i] = 0u;
}
__global__ void k_zero_f32(float* __restrict__ p, int n) {
  int i = blockIdx.x * blockDim.x + threadIdx.x;
  if (i < n) p[i] = 0.0f;
}
__global__ void k_deg_count(const long long* __restrict__ col, unsigned* __restrict__ deg,
                            long long E) {
  long long e = (long long)blockIdx.x * blockDim.x + threadIdx.x;
  if (e < E) atomicAdd(&deg[col[e]], 1u);
}
__global__ void k_dinv(const unsigned* __restrict__ deg, float* __restrict__ dinv, long long N) {
  long long i = (long long)blockIdx.x * blockDim.x + threadIdx.x;
  if (i < N) dinv[i] = rsqrtf((float)(deg[i] + 1u));  // +1 self-loop
}

// Pack W[K][N] (row-major f32) into WMMA B-fragment order:
// dst[((ks*(N/16)+nt)*32 + lane)*16 + j] = W[ks*32 + kOf(lane,j)][nt*16 + (lane&15)]
__global__ void k_pack_w(const float* __restrict__ W, half_t* __restrict__ dst, int K, int N) {
  int idx = blockIdx.x * blockDim.x + threadIdx.x;
  if (idx >= K * N) return;
  int j    = idx & 15;
  int lane = (idx >> 4) & 31;
  int frag = idx >> 9;
  int NT = N >> 4;
  int ks = frag / NT, nt = frag % NT;
  int k = ks * 32 + ((lane >= 16) ? 8 : 0) + ((j < 8) ? j : j + 8);
  int n = nt * 16 + (lane & 15);
  dst[idx] = (half_t)W[k * N + n];
}

// ---------------------------------------------------------------------------
// GCN: t = x @ W1  (K=3, per-element)
// ---------------------------------------------------------------------------
__global__ void k_xw1(const float* __restrict__ x, const float* __restrict__ W1,
                      float* __restrict__ t, long long N) {
  long long idx = (long long)blockIdx.x * blockDim.x + threadIdx.x;
  if (idx >= N * HID) return;
  long long node = idx >> 6;
  int f = (int)(idx & 63);
  t[idx] = x[node * 3 + 0] * W1[0 * HID + f]
         + x[node * 3 + 1] * W1[1 * HID + f]
         + x[node * 3 + 2] * W1[2 * HID + f];
}

// out = b + dinv[i]^2 * t[i]   (bias + self-loop message), float4 per thread
__global__ void k_gcn_init(const float* __restrict__ t, const float* __restrict__ dinv,
                           const float* __restrict__ b, float* __restrict__ out, long long N) {
  long long idx = (long long)blockIdx.x * blockDim.x + threadIdx.x;  // N*16
  if (idx >= N * 16) return;
  long long node = idx >> 4;
  int q = (int)(idx & 15);
  float d2 = dinv[node] * dinv[node];
  float4 tv = *(const float4*)(t + node * HID + q * 4);
  float4 bv = *(const float4*)(b + q * 4);
  float4 o;
  o.x = bv.x + d2 * tv.x; o.y = bv.y + d2 * tv.y;
  o.z = bv.z + d2 * tv.z; o.w = bv.w + d2 * tv.w;
  *(float4*)(out + node * HID + q * 4) = o;
}

// scatter-add: out[col] += t[row] * (dinv[row]*dinv[col]); thread per (edge, 4 feats)
__global__ void k_gcn_edge(const float* __restrict__ t, const float* __restrict__ dinv,
                           const long long* __restrict__ row, const long long* __restrict__ col,
                           float* __restrict__ out, long long E) {
  long long tid = (long long)blockIdx.x * blockDim.x + threadIdx.x;  // E*16
  if (tid >= E * 16) return;
  long long e = tid >> 4;
  int part = (int)(tid & 15);
  long long r = row[e], c = col[e];
  float w = dinv[r] * dinv[c];
  float4 v = *(const float4*)(t + r * HID + part * 4);
  float* o = out + c * HID + part * 4;
  atomicAdd(o + 0, v.x * w);
  atomicAdd(o + 1, v.y * w);
  atomicAdd(o + 2, v.z * w);
  atomicAdd(o + 3, v.w * w);
}

// ---------------------------------------------------------------------------
// WMMA node GEMM:  out[N,64] = relu?(Hin)[N,64] @ W[64,64]
// B fragments (8) held in registers; grid-stride over M-tiles.
// ---------------------------------------------------------------------------
template <int RELU>
__global__ __launch_bounds__(256)
void k_gemm_nodes(const float* __restrict__ Hin, const half_t* __restrict__ Wp,
                  float* __restrict__ out, long long ntiles) {
  int wave = threadIdx.x >> 5, lane = threadIdx.x & 31;
  int kb = (lane >= 16) ? 8 : 0;

  v16h bfrag[8];
#pragma unroll
  for (int fi = 0; fi < 8; ++fi)
    bfrag[fi] = *(const v16h*)(Wp + (size_t)(fi * 32 + lane) * 16);

  long long stride = (long long)gridDim.x * 8;
  for (long long tile = (long long)blockIdx.x * 8 + wave; tile < ntiles; tile += stride) {
    long long r0 = tile * 16 + (lane & 15);

    v8f acc[4];
#pragma unroll
    for (int nt = 0; nt < 4; ++nt)
#pragma unroll
      for (int i = 0; i < 8; ++i) acc[nt][i] = 0.0f;

#pragma unroll
    for (int ks = 0; ks < 2; ++ks) {
      const float* hr = Hin + r0 * HID + ks * 32 + kb;
      v16h a;
#pragma unroll
      for (int j = 0; j < 8; ++j) {
        float v = hr[j];
        if (RELU) v = fmaxf(v, 0.0f);
        a[j] = (half_t)v;
      }
#pragma unroll
      for (int j = 0; j < 8; ++j) {
        float v = hr[16 + j];
        if (RELU) v = fmaxf(v, 0.0f);
        a[8 + j] = (half_t)v;
      }
#pragma unroll
      for (int nt = 0; nt < 4; ++nt)
        acc[nt] = __builtin_amdgcn_wmma_f32_16x16x32_f16(false, a, false, bfrag[ks * 4 + nt],
                                                         (short)0, acc[nt], false, false);
    }
    long long mbase = tile * 16 + ((lane >= 16) ? 8 : 0);
    int n = lane & 15;
#pragma unroll
    for (int nt = 0; nt < 4; ++nt)
#pragma unroll
      for (int r = 0; r < 8; ++r)
        out[(mbase + r) * HID + nt * 16 + n] = acc[nt][r];
  }
}

// ---------------------------------------------------------------------------
// MLP layer 1: z1 = relu([h3[src] ‖ h3[dst]] @ Wm1 + bm1)
// K=128 (4 k-steps), N=64 (4 n-tiles); all 16 B fragments in registers.
// Per-feature sum/sum^2 accumulated in LDS, flushed once per block.
// ---------------------------------------------------------------------------
__global__ __launch_bounds__(256)
void k_mlp1(const float* __restrict__ h3, const long long* __restrict__ ei,
            const half_t* __restrict__ Wp, const float* __restrict__ bm1,
            half_t* __restrict__ z1, float* __restrict__ gsum, float* __restrict__ gsq,
            long long E, long long ntiles) {
  __shared__ float lsum[64], lsq[64];
  int t = threadIdx.x;
  if (t < 64) { lsum[t] = 0.0f; lsq[t] = 0.0f; }
  __syncthreads();

  int wave = t >> 5, lane = t & 31;
  int kb = (lane >= 16) ? 8 : 0;

  v16h bfrag[16];
#pragma unroll
  for (int fi = 0; fi < 16; ++fi)
    bfrag[fi] = *(const v16h*)(Wp + (size_t)(fi * 32 + lane) * 16);

  long long stride = (long long)gridDim.x * 8;
  for (long long tile = (long long)blockIdx.x * 8 + wave; tile < ntiles; tile += stride) {
    long long e = tile * 16 + (lane & 15);
    long long src = ei[e];
    long long dst = ei[E + e];

    v8f acc[4];
#pragma unroll
    for (int nt = 0; nt < 4; ++nt)
#pragma unroll
      for (int i = 0; i < 8; ++i) acc[nt][i] = 0.0f;

#pragma unroll
    for (int ks = 0; ks < 4; ++ks) {
      int koff = ks * 32 + kb;  // 0..127
      const float* base = (koff < HID) ? (h3 + src * HID + koff)
                                       : (h3 + dst * HID + (koff - HID));
      v16h a;
#pragma unroll
      for (int j = 0; j < 8; ++j) a[j] = (half_t)base[j];
#pragma unroll
      for (int j = 0; j < 8; ++j) a[8 + j] = (half_t)base[16 + j];
#pragma unroll
      for (int nt = 0; nt < 4; ++nt)
        acc[nt] = __builtin_amdgcn_wmma_f32_16x16x32_f16(false, a, false, bfrag[ks * 4 + nt],
                                                         (short)0, acc[nt], false, false);
    }
    int n = lane & 15;
    long long ebase = tile * 16 + ((lane >= 16) ? 8 : 0);
#pragma unroll
    for (int nt = 0; nt < 4; ++nt) {
      int f = nt * 16 + n;
      float bias = bm1[f];
      float ls = 0.0f, lq = 0.0f;
#pragma unroll
      for (int r = 0; r < 8; ++r) {
        float v = fmaxf(acc[nt][r] + bias, 0.0f);
        z1[(ebase + r) * HID + f] = (half_t)v;
        ls += v; lq += v * v;
      }
      atomicAdd(&lsum[f], ls);
      atomicAdd(&lsq[f], lq);
    }
  }
  __syncthreads();
  if (t < 64) { atomicAdd(&gsum[t], lsum[t]); atomicAdd(&gsq[t], lsq[t]); }
}

// BN finalize: a = g * rsqrt(var + eps), c = be - mu * a
__global__ void k_bn_fin(const float* __restrict__ sum, const float* __restrict__ sq,
                         const float* __restrict__ g, const float* __restrict__ be,
                         float* __restrict__ a, float* __restrict__ c, int F, float invE) {
  int f = threadIdx.x;
  if (f >= F) return;
  float mu  = sum[f] * invE;
  float var = sq[f] * invE - mu * mu;
  float s = g[f] * rsqrtf(var + 1e-5f);
  a[f] = s;
  c[f] = be[f] - mu * s;
}

// ---------------------------------------------------------------------------
// MLP layer 2: z2 = relu((a1*z1+c1) @ Wm2 + bm2)
// K=64 (2 k-steps), N=32 (2 n-tiles); all 4 B fragments in registers.
// ---------------------------------------------------------------------------
__global__ __launch_bounds__(256)
void k_mlp2(const half_t* __restrict__ z1, const half_t* __restrict__ Wp,
            const float* __restrict__ bm2, const float* __restrict__ a1,
            const float* __restrict__ c1, half_t* __restrict__ z2,
            float* __restrict__ gsum, float* __restrict__ gsq, long long ntiles) {
  __shared__ float lsum[32], lsq[32];
  int t = threadIdx.x;
  if (t < 32) { lsum[t] = 0.0f; lsq[t] = 0.0f; }
  __syncthreads();

  int wave = t >> 5, lane = t & 31;
  int kb = (lane >= 16) ? 8 : 0;

  v16h bfrag[4];
#pragma unroll
  for (int fi = 0; fi < 4; ++fi)
    bfrag[fi] = *(const v16h*)(Wp + (size_t)(fi * 32 + lane) * 16);

  // per-lane BN affine coefficients for the K indices this lane touches
  float aco[16], cco[16];
#pragma unroll
  for (int ks = 0; ks < 2; ++ks) {
    int k0 = ks * 32 + kb;
#pragma unroll
    for (int j = 0; j < 8; ++j) {
      aco[ks * 8 + j] = a1[k0 + j];          // first run only; the +16 run uses aco2/cco2
      cco[ks * 8 + j] = c1[k0 + j];
    }
  }
  float aco2[16], cco2[16];
#pragma unroll
  for (int ks = 0; ks < 2; ++ks) {
    int k0 = ks * 32 + kb + 16;
#pragma unroll
    for (int j = 0; j < 8; ++j) {
      aco2[ks * 8 + j] = a1[k0 + j];
      cco2[ks * 8 + j] = c1[k0 + j];
    }
  }

  long long stride = (long long)gridDim.x * 8;
  for (long long tile = (long long)blockIdx.x * 8 + wave; tile < ntiles; tile += stride) {
    long long e = tile * 16 + (lane & 15);

    v8f acc[2];
#pragma unroll
    for (int nt = 0; nt < 2; ++nt)
#pragma unroll
      for (int i = 0; i < 8; ++i) acc[nt][i] = 0.0f;

#pragma unroll
    for (int ks = 0; ks < 2; ++ks) {
      int k0 = ks * 32 + kb;
      const half_t* zr = z1 + e * HID + k0;
      v16h a;
#pragma unroll
      for (int j = 0; j < 8; ++j)
        a[j] = (half_t)(aco[ks * 8 + j] * (float)zr[j] + cco[ks * 8 + j]);
#pragma unroll
      for (int j = 0; j < 8; ++j)
        a[8 + j] = (half_t)(aco2[ks * 8 + j] * (float)zr[16 + j] + cco2[ks * 8 + j]);
#pragma unroll
      for (int nt = 0; nt < 2; ++nt)
        acc[nt] = __builtin_amdgcn_wmma_f32_16x16x32_f16(false, a, false, bfrag[ks * 2 + nt],
                                                         (short)0, acc[nt], false, false);
    }
    int n = lane & 15;
    long long ebase = tile * 16 + ((lane >= 16) ? 8 : 0);
#pragma unroll
    for (int nt = 0; nt < 2; ++nt) {
      int f = nt * 16 + n;
      float bias = bm2[f];
      float ls = 0.0f, lq = 0.0f;
#pragma unroll
      for (int r = 0; r < 8; ++r) {
        float v = fmaxf(acc[nt][r] + bias, 0.0f);
        z2[(ebase + r) * 32 + f] = (half_t)v;
        ls += v; lq += v * v;
      }
      atomicAdd(&lsum[f], ls);
      atomicAdd(&lsq[f], lq);
    }
  }
  __syncthreads();
  if (t < 32) { atomicAdd(&gsum[t], lsum[t]); atomicAdd(&gsq[t], lsq[t]); }
}

// ---------------------------------------------------------------------------
// output: out[E,2] = (a2*z2+c2) @ Wm3 + bm3   (K=32, N=2, thread per edge)
// ---------------------------------------------------------------------------
__global__ __launch_bounds__(256)
void k_out(const half_t* __restrict__ z2, const float* __restrict__ a2,
           const float* __restrict__ c2, const float* __restrict__ Wm3,
           const float* __restrict__ bm3, float* __restrict__ out, long long E) {
  __shared__ float w[64], as[32], cs[32], bs[2];
  int t = threadIdx.x;
  if (t < 64) w[t] = Wm3[t];
  if (t < 32) { as[t] = a2[t]; cs[t] = c2[t]; }
  if (t < 2) bs[t] = bm3[t];
  __syncthreads();
  long long e = (long long)blockIdx.x * blockDim.x + t;
  if (e >= E) return;
  const half_t* zr = z2 + e * 32;
  v16h zlo = *(const v16h*)(zr);
  v16h zhi = *(const v16h*)(zr + 16);
  float o0 = bs[0], o1 = bs[1];
#pragma unroll
  for (int k = 0; k < 16; ++k) {
    float v = as[k] * (float)zlo[k] + cs[k];
    o0 += v * w[k * 2 + 0];
    o1 += v * w[k * 2 + 1];
  }
#pragma unroll
  for (int k = 0; k < 16; ++k) {
    float v = as[16 + k] * (float)zhi[k] + cs[16 + k];
    o0 += v * w[(16 + k) * 2 + 0];
    o1 += v * w[(16 + k) * 2 + 1];
  }
  out[e * 2 + 0] = o0;
  out[e * 2 + 1] = o1;
}

// ---------------------------------------------------------------------------
// launch
// ---------------------------------------------------------------------------
extern "C" void kernel_launch(void* const* d_in, const int* in_sizes, int n_in,
                              void* d_out, int out_size, void* d_ws, size_t ws_size,
                              hipStream_t stream) {
  const float*     x   = (const float*)d_in[0];
  const long long* ei  = (const long long*)d_in[1];   // int64 [2,E]
  const float*     W1  = (const float*)d_in[2];
  const float*     b1  = (const float*)d_in[3];
  const float*     W2  = (const float*)d_in[4];
  const float*     b2  = (const float*)d_in[5];
  const float*     W3  = (const float*)d_in[6];
  const float*     b3  = (const float*)d_in[7];
  const float*     Wm1 = (const float*)d_in[8];
  const float*     bm1 = (const float*)d_in[9];
  const float*     g1  = (const float*)d_in[10];
  const float*     be1 = (const float*)d_in[11];
  const float*     Wm2 = (const float*)d_in[12];
  const float*     bm2 = (const float*)d_in[13];
  const float*     g2  = (const float*)d_in[14];
  const float*     be2 = (const float*)d_in[15];
  const float*     Wm3 = (const float*)d_in[16];
  const float*     bm3 = (const float*)d_in[17];

  const long long N = in_sizes[0] / 3;
  const long long E = in_sizes[1] / 2;

  // carve workspace (256B aligned)
  char* p = (char*)d_ws;
  auto take = [&](size_t bytes) -> void* {
    void* r = (void*)p;
    p += (bytes + 255) & ~(size_t)255;
    return r;
  };
  unsigned* deg  = (unsigned*)take(N * 4);
  float*    dinv = (float*)take(N * 4);
  float*    tbuf = (float*)take((size_t)N * HID * 4);
  float*    hA   = (float*)take((size_t)N * HID * 4);
  float*    hB   = (float*)take((size_t)N * HID * 4);
  half_t*   pW2  = (half_t*)take(64 * 64 * 2);
  half_t*   pW3  = (half_t*)take(64 * 64 * 2);
  half_t*   pM1  = (half_t*)take(128 * 64 * 2);
  half_t*   pM2  = (half_t*)take(64 * 32 * 2);
  float*    st   = (float*)take(384 * 4);  // sum1|sq1|sum2|sq2|a1|c1|a2|c2
  half_t*   z1   = (half_t*)take((size_t)E * HID * 2);
  half_t*   z2   = (half_t*)take((size_t)E * 32 * 2);

  const int T = 256;
  auto gb = [](long long n, int t) { return (unsigned)((n + t - 1) / t); };

  // degrees / norm / stat init / weight packs
  k_zero_u32<<<gb(N, T), T, 0, stream>>>(deg, N);
  k_zero_f32<<<1, T, 0, stream>>>(st, 192);
  k_deg_count<<<gb(E, T), T, 0, stream>>>(ei + E, deg, E);
  k_dinv<<<gb(N, T), T, 0, stream>>>(deg, dinv, N);
  k_pack_w<<<gb(64 * 64, T), T, 0, stream>>>(W2, pW2, 64, 64);
  k_pack_w<<<gb(64 * 64, T), T, 0, stream>>>(W3, pW3, 64, 64);
  k_pack_w<<<gb(128 * 64, T), T, 0, stream>>>(Wm1, pM1, 128, 64);
  k_pack_w<<<gb(64 * 32, T), T, 0, stream>>>(Wm2, pM2, 64, 32);

  const long long ntN = N / 16;  // node M-tiles (N multiple of 16)
  const long long ntE = E / 16;  // edge M-tiles (E multiple of 16)
  // grid-stride grids: enough waves to fill the chip, small enough to reuse
  // register-resident weight fragments across several tiles per wave
  const unsigned gNodes = (unsigned)min((long long)512, (ntN + 7) / 8);
  const unsigned gEdges = (unsigned)min((long long)2048, (ntE + 7) / 8);

  // ---- GCN layer 1 ----
  k_xw1<<<gb(N * HID, T), T, 0, stream>>>(x, W1, tbuf, N);
  k_gcn_init<<<gb(N * 16, T), T, 0, stream>>>(tbuf, dinv, b1, hA, N);
  k_gcn_edge<<<gb(E * 16, T), T, 0, stream>>>(tbuf, dinv, ei, ei + E, hA, E);
  // ---- GCN layer 2 ----
  k_gemm_nodes<1><<<gNodes, T, 0, stream>>>(hA, pW2, tbuf, ntN);
  k_gcn_init<<<gb(N * 16, T), T, 0, stream>>>(tbuf, dinv, b2, hB, N);
  k_gcn_edge<<<gb(E * 16, T), T, 0, stream>>>(tbuf, dinv, ei, ei + E, hB, E);
  // ---- GCN layer 3 ----
  k_gemm_nodes<1><<<gNodes, T, 0, stream>>>(hB, pW3, tbuf, ntN);
  k_gcn_init<<<gb(N * 16, T), T, 0, stream>>>(tbuf, dinv, b3, hA, N);
  k_gcn_edge<<<gb(E * 16, T), T, 0, stream>>>(tbuf, dinv, ei, ei + E, hA, E);  // hA = h3

  // ---- edge MLP ----
  const float invE = 1.0f / (float)E;
  k_mlp1<<<gEdges, T, 0, stream>>>(hA, ei, pM1, bm1, z1, st + 0, st + 64, E, ntE);
  k_bn_fin<<<1, 64, 0, stream>>>(st + 0, st + 64, g1, be1, st + 192, st + 256, 64, invE);
  k_mlp2<<<gEdges, T, 0, stream>>>(z1, pM2, bm2, st + 192, st + 256, z2, st + 128, st + 160, ntE);
  k_bn_fin<<<1, 64, 0, stream>>>(st + 128, st + 160, g2, be2, st + 320, st + 352, 32, invE);
  k_out<<<gb(E, T), T, 0, stream>>>(z2, st + 320, st + 352, Wm3, bm3, (float*)d_out, E);
}
